// WeightOnlyLinear_47339129537060
// MI455X (gfx1250) — matrix operand order repaired and checked by reading, not compiled, not run on hardware
//
#include <hip/hip_runtime.h>

// ---- WMMA / TDM vector types (gfx1250, wave32) ----
typedef __attribute__((ext_vector_type(16))) __bf16 v16bf;
typedef __attribute__((ext_vector_type(8)))  float  v8f;
typedef __attribute__((ext_vector_type(4)))  unsigned int u32x4;
typedef __attribute__((ext_vector_type(8)))  int    i32x8;
typedef __attribute__((ext_vector_type(4)))  int    i32x4;

#define TILE_M 128
#define TILE_N 128
#define TILE_K 64          // == quant BLOCK: one (scale, zp) pair per K-chunk
#define LDS_PITCH 72       // bf16 elems/row: 144B = 36 dwords -> 16B aligned + conflict-free

union Frag32B {            // 32 bytes: one 16x32 (A) or 32x16 (B) bf16 fragment per lane
    uint4 u[2];
    v16bf v;
};

__device__ __forceinline__ unsigned int f32_to_bf16_bits(float f) {
    unsigned int u = __float_as_uint(f);
    u += 0x7FFFu + ((u >> 16) & 1u);   // round-to-nearest-even
    return u >> 16;
}

// ---------------------------------------------------------------------------
// Pre-pass: convert x (f32) -> bf16 once into workspace (removes the per-K-iter
// conversion that would otherwise be repeated by all 86 N-tile blocks).
// ---------------------------------------------------------------------------
__global__ __launch_bounds__(256)
void convert_x_to_bf16(const float* __restrict__ x,
                       unsigned short* __restrict__ xh, long long n) {
    long long i = ((long long)blockIdx.x * 256 + threadIdx.x) * 8;
    if (i + 8 <= n) {
        float4 a = *(const float4*)(x + i);
        float4 b = *(const float4*)(x + i + 4);
        uint4 p;
        p.x = f32_to_bf16_bits(a.x) | (f32_to_bf16_bits(a.y) << 16);
        p.y = f32_to_bf16_bits(a.z) | (f32_to_bf16_bits(a.w) << 16);
        p.z = f32_to_bf16_bits(b.x) | (f32_to_bf16_bits(b.y) << 16);
        p.w = f32_to_bf16_bits(b.z) | (f32_to_bf16_bits(b.w) << 16);
        *(uint4*)(xh + i) = p;
    } else {
        for (; i < n; ++i) xh[i] = (unsigned short)f32_to_bf16_bits(x[i]);
    }
}

// ---------------------------------------------------------------------------
// Main kernel: double-buffered LDS, TDM A-tile copy (PRECONV) or inline
// conversion fallback, int4 dequant pipelined behind the WMMA block.
// ---------------------------------------------------------------------------
template <bool PRECONV>
__global__ __launch_bounds__(256, 2)
void wqlinear_int4_bf16_wmma(const float* __restrict__ x,
                             const unsigned short* __restrict__ xh,
                             const int*   __restrict__ qweight,
                             const int*   __restrict__ qzeros,
                             const float* __restrict__ scales,
                             const float* __restrict__ bias,
                             float* __restrict__ out,
                             int tokens, int in_f, int out_f)
{
    __shared__ alignas(16) unsigned short lA[2][TILE_M * LDS_PITCH]; // x tiles
    __shared__ alignas(16) unsigned short lB[2][TILE_N * LDS_PITCH]; // W tiles

    const int tid  = threadIdx.x;
    const int lane = tid & 31;
    const int wave = tid >> 5;

    const int n0 = blockIdx.x * TILE_N;   // output-feature tile base
    const int m0 = blockIdx.y * TILE_M;   // token tile base

    // wave sub-tile: 64 (M) x 32 (N)
    const int wm = (wave & 1) * 64;
    const int wn = (wave >> 1) * 32;

    v8f acc[4][2];
    #pragma unroll
    for (int i = 0; i < 4; ++i)
        #pragma unroll
        for (int j = 0; j < 2; ++j)
            acc[i][j] = (v8f){0.f,0.f,0.f,0.f,0.f,0.f,0.f,0.f};

    const int qw_pitch = in_f >> 3;   // int32 words per weight row
    const int groups   = in_f >> 6;   // quant groups per row
    const int qz_pitch = groups >> 3; // int32 words per qzeros row

    // B-staging role of this thread (fixed across K loop)
    const int bn   = tid >> 1;        // 0..127 : output feature within tile
    const int bhal = tid & 1;         // 0/1    : K half (32 elements each)
    const int gn   = n0 + bn;
    const bool bok = (gn < out_f);

    // ---- TDM issue: DMA 128 rows x 64 bf16 (row stride in_f) into lA[buf],
    //      LDS pad 16B per 128B row -> LDS_PITCH. Issued by wave 0 only. ----
    auto issue_tdm = [&](int k0, int buf) {
        const unsigned long long ga =
            (unsigned long long)(const void*)(xh + (size_t)m0 * in_f + k0);
        u32x4 g0;
        g0[0] = 1u;                                          // count=1
        g0[1] = (unsigned)(size_t)(void*)&lA[buf][0];        // lds_addr
        g0[2] = (unsigned)(ga & 0xFFFFFFFFull);
        g0[3] = (unsigned)((ga >> 32) & 0x1FFFFFFull) | (2u << 30); // type=2
        unsigned td1 = (unsigned)(tokens - m0);
        if (td1 > 0xFFFFu) td1 = 0xFFFFu;
        i32x8 g1;
        g1[0] = (int)((1u << 16)      // data_size = 2 bytes
                    | (1u << 20)      // pad_enable
                    | (4u << 22)      // pad_interval: 32 dwords (128B)
                    | (3u << 25));    // pad_amount: 4 dwords (16B)
        g1[1] = (int)(((unsigned)in_f & 0xFFFFu) << 16);     // tensor_dim0 lo
        g1[2] = (int)((((unsigned)in_f >> 16) & 0xFFFFu) | (td1 << 16));
        g1[3] = (int)(64u << 16);     // tile_dim0 = 64
        g1[4] = 128;                  // tile_dim1 = 128, tile_dim2 = 0
        g1[5] = in_f;                 // tensor_dim0_stride
        g1[6] = 0;
        g1[7] = 0;
        i32x4 gz = (i32x4){0, 0, 0, 0};
#if defined(__clang_major__) && (__clang_major__ >= 23)
        i32x8 gz8 = (i32x8){0, 0, 0, 0, 0, 0, 0, 0};
        __builtin_amdgcn_tensor_load_to_lds(g0, g1, gz, gz, gz8, 0);
#else
        __builtin_amdgcn_tensor_load_to_lds(g0, g1, gz, gz, 0);
#endif
    };

    // ---- fallback A staging: load f32, convert, store bf16 to lA[buf] ----
    auto stage_A_inline = [&](int k0, int buf) {
        #pragma unroll
        for (int i = 0; i < 8; ++i) {
            const int idx = tid + i * 256;          // 2048 float4 slots
            const int r   = idx >> 4;
            const int c4  = idx & 15;
            float4 v = make_float4(0.f, 0.f, 0.f, 0.f);
            if (m0 + r < tokens)
                v = *(const float4*)(x + (size_t)(m0 + r) * in_f + k0 + c4 * 4);
            unsigned int p0 = f32_to_bf16_bits(v.x) | (f32_to_bf16_bits(v.y) << 16);
            unsigned int p1 = f32_to_bf16_bits(v.z) | (f32_to_bf16_bits(v.w) << 16);
            *(uint2*)&lA[buf][r * LDS_PITCH + c4 * 4] = make_uint2(p0, p1);
        }
    };

    // ---- B stage, phase 1: global loads into registers (issue early) ----
    auto load_B = [&](int k0, float& sc, float& zp, int4& q) {
        sc = 0.f; zp = 0.f; q = make_int4(0, 0, 0, 0);
        if (bok) {
            const int g = k0 >> 6;
            sc = scales[(size_t)gn * groups + g];
            const int zw = qzeros[(size_t)gn * qz_pitch + (g >> 3)];
            zp = (float)((zw >> ((g & 7) * 4)) & 0xF);
            const int* qp = qweight + (size_t)gn * qw_pitch + (k0 >> 3) + bhal * 4;
            q = *(const int4*)qp;
            if (k0 + TILE_K < in_f)
                __builtin_prefetch(qp + 8, 0, 1);   // global_prefetch_b8
        }
    };

    // ---- B stage, phase 2: dequant registers -> bf16 LDS (after compute) ----
    auto store_B = [&](int buf, float sc, float zp, int4 q) {
        const int w32[4] = {q.x, q.y, q.z, q.w};
        unsigned short* dst = &lB[buf][bn * LDS_PITCH + bhal * 32];
        #pragma unroll
        for (int w = 0; w < 4; ++w) {
            unsigned int pk[4];
            const int word = w32[w];
            #pragma unroll
            for (int j = 0; j < 4; ++j) {
                const float f0 = ((float)((word >> (8 * j))     & 0xF) - zp) * sc;
                const float f1 = ((float)((word >> (8 * j + 4)) & 0xF) - zp) * sc;
                pk[j] = f32_to_bf16_bits(f0) | (f32_to_bf16_bits(f1) << 16);
            }
            *(uint4*)(dst + w * 8) = make_uint4(pk[0], pk[1], pk[2], pk[3]);
        }
    };

    // ---- compute: 2 x (4 A-frags x 2 B-frags) WMMA from lA/lB[buf] ----
    auto compute = [&](int buf) {
        const unsigned short* As = &lA[buf][0];
        const unsigned short* Bs = &lB[buf][0];
        #pragma unroll
        for (int kk = 0; kk < TILE_K; kk += 32) {
            Frag32B aF[4];
            const int arow_in = lane & 15;
            const int klo     = kk + ((lane < 16) ? 0 : 8);
            #pragma unroll
            for (int mt = 0; mt < 4; ++mt) {
                const int row = wm + mt * 16 + arow_in;
                aF[mt].u[0] = *(const uint4*)&As[row * LDS_PITCH + klo];
                aF[mt].u[1] = *(const uint4*)&As[row * LDS_PITCH + klo + 16];
            }
            Frag32B bF[2];
            const int kb = kk + ((lane < 16) ? 0 : 8);
            #pragma unroll
            for (int nt = 0; nt < 2; ++nt) {
                const int col = wn + nt * 16 + (lane & 15);
                bF[nt].u[0] = *(const uint4*)&Bs[col * LDS_PITCH + kb];
                bF[nt].u[1] = *(const uint4*)&Bs[col * LDS_PITCH + kb + 16];
            }
            #pragma unroll
            for (int mt = 0; mt < 4; ++mt)
                #pragma unroll
                for (int nt = 0; nt < 2; ++nt)
                    acc[mt][nt] = __builtin_amdgcn_wmma_f32_16x16x32_bf16(
                        false, aF[mt].v, false, bF[nt].v,
                        (short)0, acc[mt][nt], false, false);
        }
    };

    // ---------------- prologue: stage tile 0 into buffer 0 ----------------
    if (PRECONV) {
        if (wave == 0) issue_tdm(0, 0);
    } else {
        stage_A_inline(0, 0);
    }
    {
        float sc, zp; int4 q;
        load_B(0, sc, zp, q);
        store_B(0, sc, zp, q);
    }
    if (PRECONV && wave == 0)
        __builtin_amdgcn_s_wait_tensorcnt(0);
    __syncthreads();

    // ---------------- pipelined main loop: one barrier per iteration -------
    int cur = 0;
    for (int k0 = 0; k0 < in_f; k0 += TILE_K) {
        const int  nxt      = cur ^ 1;
        const bool has_next = (k0 + TILE_K) < in_f;

        float sc = 0.f, zp = 0.f;
        int4  q  = make_int4(0, 0, 0, 0);
        if (has_next) {
            if (PRECONV) {
                if (wave == 0) issue_tdm(k0 + TILE_K, nxt);  // DMA overlaps compute
            } else {
                stage_A_inline(k0 + TILE_K, nxt);
            }
            load_B(k0 + TILE_K, sc, zp, q);   // global loads in flight over compute
        }

        compute(cur);                          // 16 WMMAs on current buffers

        if (has_next) {
            store_B(nxt, sc, zp, q);           // dequant after WMMAs issued
            if (PRECONV && wave == 0)
                __builtin_amdgcn_s_wait_tensorcnt(0);
        }
        __syncthreads();
        cur = nxt;
    }

    // ---------------- epilogue: C layout (VGPR i -> M=i / i+8), add bias ----
    const int cn_in = lane & 15;
    const int mhalf = (lane < 16) ? 0 : 8;
    #pragma unroll
    for (int mt = 0; mt < 4; ++mt) {
        #pragma unroll
        for (int nt = 0; nt < 2; ++nt) {
            const int col = n0 + wn + nt * 16 + cn_in;
            if (col >= out_f) continue;
            const float bv = bias[col];
            #pragma unroll
            for (int i = 0; i < 8; ++i) {
                const int row = m0 + wm + mt * 16 + mhalf + i;
                if (row < tokens)
                    out[(size_t)row * out_f + col] = acc[mt][nt][i] + bv;
            }
        }
    }
}

extern "C" void kernel_launch(void* const* d_in, const int* in_sizes, int n_in,
                              void* d_out, int out_size, void* d_ws, size_t ws_size,
                              hipStream_t stream) {
    const float* x       = (const float*)d_in[0];
    const int*   qweight = (const int*)  d_in[1];
    const int*   qzeros  = (const int*)  d_in[2];
    const float* scales  = (const float*)d_in[3];
    const float* bias    = (const float*)d_in[4];
    float*       out     = (float*)      d_out;

    const int out_f  = in_sizes[4];                 // 11008
    const int groups = in_sizes[3] / out_f;         // 64
    const int in_f   = groups * 64;                 // 4096
    const int tokens = in_sizes[0] / in_f;          // 4096

    const long long nx = (long long)tokens * in_f;
    const bool use_ws = (ws_size >= (size_t)nx * sizeof(unsigned short));

    dim3 grid((out_f + TILE_N - 1) / TILE_N, (tokens + TILE_M - 1) / TILE_M);

    if (use_ws) {
        unsigned short* xh = (unsigned short*)d_ws;
        const long long nblk = (nx / 8 + 255) / 256;
        convert_x_to_bf16<<<(unsigned)nblk, 256, 0, stream>>>(x, xh, nx);
        wqlinear_int4_bf16_wmma<true><<<grid, 256, 0, stream>>>(
            x, xh, qweight, qzeros, scales, bias, out, tokens, in_f, out_f);
    } else {
        wqlinear_int4_bf16_wmma<false><<<grid, 256, 0, stream>>>(
            x, (const unsigned short*)nullptr, qweight, qzeros, scales, bias, out,
            tokens, in_f, out_f);
    }
}
